// Decoder_87153476370868
// MI455X (gfx1250) — matrix operand order repaired and checked by reading, not compiled, not run on hardware
//
#include <hip/hip_runtime.h>

// ---------------------------------------------------------------------------
// MI455X (gfx1250) decoder: bf16 WMMA for LSTM GEMMs + kh/vh precompute,
// L2-resident bf16 K/V for the per-step attention GEMV,
// GLOBAL_LOAD_ASYNC_TO_LDS_B128 for GEMM A-tile staging.
// Wave32 WMMA A/B fragments use contiguous k order (k = 16*(lane/16) + s).
// ---------------------------------------------------------------------------

#define B_   128
#define T_   1024
#define D_   128
#define V_   30
#define E_   256
#define H1_  1024
#define NH_  8
#define HD_  16
#define MAXLEN_ 400

typedef __attribute__((ext_vector_type(16))) __bf16 v16bf;
typedef __attribute__((ext_vector_type(8)))  float  v8f;

// ---- bf16 bit helpers (RNE) ----
__device__ __forceinline__ unsigned short f2bfbits(float f) {
    union { float f; unsigned u; } a; a.f = f;
    unsigned u = a.u;
    unsigned r = u + 0x7FFFu + ((u >> 16) & 1u);
    return (unsigned short)(r >> 16);
}
__device__ __forceinline__ float bfbits2f(unsigned short h) {
    union { unsigned u; float f; } a; a.u = ((unsigned)h) << 16;
    return a.f;
}
__device__ __forceinline__ __bf16 bits2bf(unsigned short h) {
    __bf16 b; __builtin_memcpy(&b, &h, 2); return b;
}
__device__ __forceinline__ float sigmoidf_(float x) {
    return 1.0f / (1.0f + __expf(-x));
}

// ---- CDNA5 async global->LDS copy (ASYNCcnt-tracked, bypasses VGPRs) ----
// Generic LDS addresses carry the LDS byte offset in the low 32 bits
// (ISA 10.2 aperture model), so truncation gives the VDST LDS address.
__device__ __forceinline__ void async_ld_b128(void* lds, const void* gsrc) {
    unsigned loff = (unsigned)(size_t)lds;
    asm volatile("global_load_async_to_lds_b128 %0, %1, off"
                 :: "v"(loff), "v"(gsrc) : "memory");
}
__device__ __forceinline__ void wait_async0() {
    asm volatile("s_wait_asynccnt 0x0" ::: "memory");
}

// ---- WMMA bf16 16x16x32, fp32 accumulate ----
__device__ __forceinline__ v8f wmma_bf(v16bf a, v16bf b, v8f c) {
    return __builtin_amdgcn_wmma_f32_16x16x32_bf16(
        /*neg_a=*/false, a, /*neg_b=*/false, b,
        /*c_mod=*/(short)0, c, /*reuse_a=*/false, /*reuse_b=*/false);
}

// A fragment: 16x32 (MxK), row-major bf16-bits source, row stride lda.
// Layout: lanes 0-15 -> M=lane, K=0..15 ; lanes 16-31 -> K=16..31 (matches B).
__device__ __forceinline__ v16bf load_a_bf(const unsigned short* A, int lda, int lane) {
    int m  = lane & 15;
    int kb = (lane & 16) ? 16 : 0;
    const unsigned short* row = A + (size_t)m * lda;
    v16bf a;
#pragma unroll
    for (int j = 0; j < 16; ++j) {
        a[j]     = bits2bf(row[kb + j]);
    }
    return a;
}

// B fragment: 32x16 (KxN) where Bmat[k,n] = W[n,k], W row-major with stride ldw.
// Layout (contiguous k, matches A): lanes 0-15 -> N=lane, K=0..15 ; lanes 16-31 -> K=16..31.
__device__ __forceinline__ v16bf load_b_bf(const unsigned short* W, int ldw, int lane) {
    int n  = lane & 15;
    int kb = (lane & 16) ? 16 : 0;
    const unsigned short* row = W + (size_t)n * ldw + kb;
    v16bf b;
#pragma unroll
    for (int j = 0; j < 16; ++j) b[j] = bits2bf(row[j]);
    return b;
}

// ---------------------------------------------------------------------------
// Prep: zero preds, h2 <- hidden0, argmax <- 0
// ---------------------------------------------------------------------------
__global__ void init_kernel(float* __restrict__ preds, int n_preds,
                            const float* __restrict__ hidden0,
                            float* __restrict__ h2f, int* __restrict__ amax) {
    int i = blockIdx.x * blockDim.x + threadIdx.x;
    int stride = gridDim.x * blockDim.x;
    for (int j = i; j < n_preds; j += stride) preds[j] = 0.0f;
    for (int j = i; j < B_ * D_;  j += stride) h2f[j] = hidden0[j];
    for (int j = i; j < B_;       j += stride) amax[j] = 0;
}

__global__ void cvt_kernel(const float* __restrict__ src,
                           unsigned short* __restrict__ dst, int n) {
    int i = blockIdx.x * blockDim.x + threadIdx.x;
    int stride = gridDim.x * blockDim.x;
    for (int j = i; j < n; j += stride) dst[j] = f2bfbits(src[j]);
}

// ---------------------------------------------------------------------------
// kh/vh precompute: out[b][h][t][d] (bf16) = X[b,t,:] @ W^T + bias   (WMMA)
// M = B*T (8192 tiles of 16), K = 128, N = 128 (n-tile == head).
// Block = 256 threads = 8 waves; wave w handles head w; A tile staged in LDS.
// Two accumulators break the D->C WMMA dependency chain (no hazard NOPs).
// ---------------------------------------------------------------------------
__global__ void khvh_kernel(const float* __restrict__ X,            // [B*T,128]
                            const unsigned short* __restrict__ Wbf, // [128,128]
                            const float* __restrict__ bias,         // [128]
                            unsigned short* __restrict__ out) {     // [B][8][T][16]
    __shared__ __align__(16) unsigned short sA[16 * 128];
    int m0  = blockIdx.x * 16;
    int tid = threadIdx.x;
    for (int idx = tid; idx < 16 * 128; idx += 256) {
        int r = idx >> 7, c = idx & 127;
        sA[idx] = f2bfbits(X[(size_t)(m0 + r) * 128 + c]);
    }
    __syncthreads();
    int wave = tid >> 5, lane = tid & 31;
    int n0 = wave * 16;                    // head = wave
    v8f c0 = {}, c1 = {};
#pragma unroll
    for (int k0 = 0; k0 < 64; k0 += 32) {
        v16bf a = load_a_bf(sA + k0, 128, lane);
        v16bf b = load_b_bf(Wbf + (size_t)n0 * 128 + k0, 128, lane);
        c0 = wmma_bf(a, b, c0);
        v16bf a2 = load_a_bf(sA + 64 + k0, 128, lane);
        v16bf b2 = load_b_bf(Wbf + (size_t)n0 * 128 + 64 + k0, 128, lane);
        c1 = wmma_bf(a2, b2, c1);
    }
    int nloc = lane & 15;
    float bv = bias[n0 + nloc];
    int rbase = (lane & 16) ? 8 : 0;
#pragma unroll
    for (int r = 0; r < 8; ++r) {
        int m  = m0 + rbase + r;
        int b_ = m >> 10;                  // / T
        int t  = m & 1023;
        out[(((size_t)b_ * NH_ + wave) * T_ + t) * HD_ + nloc] =
            f2bfbits(c0[r] + c1[r] + bv);
    }
}

// ---------------------------------------------------------------------------
// Attention: one block per (b,h); q = h2 @ Wq^T + bq computed inline (fp32);
// masked softmax over T; ctx accumulated. K/V read bf16 from L2-resident ws.
// ---------------------------------------------------------------------------
__global__ void att_kernel(const float* __restrict__ h2f,           // [B,128]
                           const float* __restrict__ Wq,            // [128,128]
                           const float* __restrict__ bq,            // [128]
                           const unsigned short* __restrict__ khL,  // [B][8][T][16]
                           const unsigned short* __restrict__ vhL,
                           const int* __restrict__ enc_len,
                           float* __restrict__ ctxf,                // [B,128]
                           unsigned short* __restrict__ x1bf) {     // [B,384]
    __shared__ float q[16];
    __shared__ float sc[T_];
    __shared__ float red[128];
    __shared__ float cacc[128][16];
    int bh = blockIdx.x;
    int b = bh >> 3, h = bh & 7;
    int tid = threadIdx.x;                 // 0..127
    if (tid < 16) {
        float acc = bq[h * 16 + tid];
        const float* wrow = Wq + (size_t)(h * 16 + tid) * 128;
        const float* hrow = h2f + (size_t)b * 128;
        for (int e = 0; e < 128; ++e) acc += hrow[e] * wrow[e];
        q[tid] = acc;
    }
    __syncthreads();
    int len = enc_len[b];
    const unsigned short* K  = khL + (size_t)bh * T_ * HD_;
    const unsigned short* Vv = vhL + (size_t)bh * T_ * HD_;
    float mx = -3.4e38f;
    for (int t = tid; t < T_; t += 128) {
        if (t + 128 < T_) __builtin_prefetch(K + (t + 128) * HD_, 0, 1);
        float s = -3.4e38f;
        if (t < len) {
            const unsigned short* kr = K + t * HD_;
            float acc = 0.0f;
#pragma unroll
            for (int d = 0; d < 16; ++d) acc += q[d] * bfbits2f(kr[d]);
            s = acc * 0.25f;               // 1/sqrt(16)
        }
        sc[t] = s;
        mx = fmaxf(mx, s);
    }
    red[tid] = mx;
    __syncthreads();
    for (int off = 64; off > 0; off >>= 1) {
        if (tid < off) red[tid] = fmaxf(red[tid], red[tid + off]);
        __syncthreads();
    }
    mx = red[0];
    __syncthreads();
    float sum = 0.0f, acc16[16];
#pragma unroll
    for (int d = 0; d < 16; ++d) acc16[d] = 0.0f;
    for (int t = tid; t < T_; t += 128) {
        if (t + 128 < T_) __builtin_prefetch(Vv + (t + 128) * HD_, 0, 1);
        float e = (sc[t] > -3.0e38f) ? __expf(sc[t] - mx) : 0.0f;
        sum += e;
        const unsigned short* vr = Vv + t * HD_;
#pragma unroll
        for (int d = 0; d < 16; ++d) acc16[d] += e * bfbits2f(vr[d]);
    }
    red[tid] = sum;
#pragma unroll
    for (int d = 0; d < 16; ++d) cacc[tid][d] = acc16[d];
    __syncthreads();
    for (int off = 64; off > 0; off >>= 1) {
        if (tid < off) {
            red[tid] += red[tid + off];
#pragma unroll
            for (int d = 0; d < 16; ++d) cacc[tid][d] += cacc[tid + off][d];
        }
        __syncthreads();
    }
    if (tid < 16) {
        float c = cacc[0][tid] / red[0];
        ctxf[(size_t)b * 128 + h * 16 + tid]  = c;
        x1bf[(size_t)b * 384 + h * 16 + tid]  = f2bfbits(c);
    }
}

// ---------------------------------------------------------------------------
// Token select (teacher forcing vs greedy) + embedding gather into x1[:,128:384]
// ---------------------------------------------------------------------------
__global__ void tok_kernel(const int* __restrict__ tf_mask, const int* __restrict__ y,
                           int step, const int* __restrict__ amax,
                           const unsigned short* __restrict__ embbf,
                           unsigned short* __restrict__ x1bf) {
    int b = blockIdx.x;
    __shared__ int tok;
    if (threadIdx.x == 0)
        tok = tf_mask[step] ? y[b * MAXLEN_ + step] : amax[b];
    __syncthreads();
    const unsigned short* src = embbf + (size_t)tok * E_;
    unsigned short* dst = x1bf + (size_t)b * 384 + 128;
    for (int j = threadIdx.x; j < E_; j += blockDim.x) dst[j] = src[j];
}

// ---------------------------------------------------------------------------
// gates1 + LSTM1 (zero state => forget gate dead, skipped):
// h1 = sigmoid(o) * tanh(sigmoid(i) * tanh(g));  gates = x1 @ W_ih1^T + b1
// Grid (8 m-tiles, 8 n-groups), block 256 = 8 waves; wave -> one 16-col n-tile,
// 3 gate accumulators, K = 384. A tile staged via async global->LDS DMA.
// ---------------------------------------------------------------------------
__global__ void g1_kernel(const unsigned short* __restrict__ x1bf,  // [128,384]
                          const unsigned short* __restrict__ W1bf,  // [4096,384]
                          const float* __restrict__ b1,             // [4096]
                          unsigned short* __restrict__ h1bf) {      // [128,1024]
    __shared__ __align__(16) unsigned short sA[16 * 384];
    int m0  = blockIdx.x * 16;
    int tid = threadIdx.x;
    {   // 16 rows x 384 halfs, rows contiguous in global -> 768 x b128 async copies
        const unsigned short* gsrc = x1bf + (size_t)m0 * 384;
        for (int idx = tid; idx < 768; idx += 256)
            async_ld_b128(&sA[idx * 8], gsrc + idx * 8);
        wait_async0();
    }
    __syncthreads();
    int wave = tid >> 5, lane = tid & 31;
    int n0 = (blockIdx.y * 8 + wave) * 16;           // 0..1008
    v8f ci = {}, cg = {}, co = {};
    for (int k0 = 0; k0 < 384; k0 += 32) {
        v16bf a  = load_a_bf(sA + k0, 384, lane);
        v16bf bi = load_b_bf(W1bf + (size_t)(0 * H1_ + n0) * 384 + k0, 384, lane);
        v16bf bg = load_b_bf(W1bf + (size_t)(2 * H1_ + n0) * 384 + k0, 384, lane);
        v16bf bo = load_b_bf(W1bf + (size_t)(3 * H1_ + n0) * 384 + k0, 384, lane);
        ci = wmma_bf(a, bi, ci);
        cg = wmma_bf(a, bg, cg);
        co = wmma_bf(a, bo, co);
    }
    int nloc = lane & 15, n = n0 + nloc;
    float bi_ = b1[n], bg_ = b1[2 * H1_ + n], bo_ = b1[3 * H1_ + n];
    int rbase = (lane & 16) ? 8 : 0;
#pragma unroll
    for (int r = 0; r < 8; ++r) {
        float iv = sigmoidf_(ci[r] + bi_);
        float gv = tanhf(cg[r] + bg_);
        float ov = sigmoidf_(co[r] + bo_);
        h1bf[(size_t)(m0 + rbase + r) * H1_ + n] = f2bfbits(ov * tanhf(iv * gv));
    }
}

// ---------------------------------------------------------------------------
// gates2 + LSTM2: h2 = lstm(h1 @ W_ih2^T + b2), forget gate skipped.
// Grid 8 m-tiles, block 256 = 8 waves (wave -> n-tile of D=128), K = 1024.
// 32KB A tile staged via async global->LDS DMA.
// ---------------------------------------------------------------------------
__global__ void g2_kernel(const unsigned short* __restrict__ h1bf,  // [128,1024]
                          const unsigned short* __restrict__ W2bf,  // [512,1024]
                          const float* __restrict__ b2,             // [512]
                          float* __restrict__ h2f) {                // [128,128]
    __shared__ __align__(16) unsigned short sA[16 * 1024];
    int m0  = blockIdx.x * 16;
    int tid = threadIdx.x;
    {   // 16 rows x 1024 halfs contiguous -> 2048 x b128 async copies
        const unsigned short* gsrc = h1bf + (size_t)m0 * H1_;
        for (int idx = tid; idx < 2048; idx += 256)
            async_ld_b128(&sA[idx * 8], gsrc + idx * 8);
        wait_async0();
    }
    __syncthreads();
    int wave = tid >> 5, lane = tid & 31;
    int n0 = wave * 16;
    v8f ci = {}, cg = {}, co = {};
    for (int k0 = 0; k0 < 1024; k0 += 32) {
        v16bf a  = load_a_bf(sA + k0, 1024, lane);
        v16bf bi = load_b_bf(W2bf + (size_t)(0 * D_ + n0) * H1_ + k0, H1_, lane);
        v16bf bg = load_b_bf(W2bf + (size_t)(2 * D_ + n0) * H1_ + k0, H1_, lane);
        v16bf bo = load_b_bf(W2bf + (size_t)(3 * D_ + n0) * H1_ + k0, H1_, lane);
        ci = wmma_bf(a, bi, ci);
        cg = wmma_bf(a, bg, cg);
        co = wmma_bf(a, bo, co);
    }
    int nloc = lane & 15, n = n0 + nloc;
    float bi_ = b2[n], bg_ = b2[2 * D_ + n], bo_ = b2[3 * D_ + n];
    int rbase = (lane & 16) ? 8 : 0;
#pragma unroll
    for (int r = 0; r < 8; ++r) {
        float iv = sigmoidf_(ci[r] + bi_);
        float gv = tanhf(cg[r] + bg_);
        float ov = sigmoidf_(co[r] + bo_);
        h2f[(size_t)(m0 + rbase + r) * D_ + n] = ov * tanhf(iv * gv);
    }
}

// ---------------------------------------------------------------------------
// logits = [h2, ctx] @ emb^T + b_out ; out = elu ; store preds slot ; argmax
// ---------------------------------------------------------------------------
__global__ void out_kernel(const float* __restrict__ h2f, const float* __restrict__ ctxf,
                           const float* __restrict__ emb, const float* __restrict__ b_out,
                           float* __restrict__ preds, int slot, int* __restrict__ amax) {
    int b = blockIdx.x;
    int v = threadIdx.x;                   // 32 threads
    __shared__ float z[32];
    float val = -3.4e38f;
    if (v < V_) {
        float acc = b_out[v];
        const float* er = emb + (size_t)v * E_;
        const float* hr = h2f + (size_t)b * D_;
        const float* cr = ctxf + (size_t)b * D_;
        for (int e = 0; e < 128; ++e) acc += hr[e] * er[e] + cr[e] * er[128 + e];
        val = acc;
        float o = acc > 0.0f ? acc : (__expf(acc) - 1.0f);
        preds[((size_t)b * V_ + v) * MAXLEN_ + slot] = o;
    }
    z[v] = val;
    __syncthreads();
    if (v == 0) {
        int best = 0; float bv = z[0];
        for (int j = 1; j < V_; ++j) if (z[j] > bv) { bv = z[j]; best = j; }
        amax[b] = best;
    }
}

// ---------------------------------------------------------------------------
extern "C" void kernel_launch(void* const* d_in, const int* in_sizes, int n_in,
                              void* d_out, int out_size, void* d_ws, size_t ws_size,
                              hipStream_t stream) {
    (void)in_sizes; (void)n_in; (void)out_size; (void)ws_size;
    const float* key     = (const float*)d_in[0];
    const float* value   = (const float*)d_in[1];
    const float* emb     = (const float*)d_in[2];
    const float* Wq      = (const float*)d_in[3];
    const float* bq      = (const float*)d_in[4];
    const float* Wk      = (const float*)d_in[5];
    const float* bk      = (const float*)d_in[6];
    const float* Wv      = (const float*)d_in[7];
    const float* bv      = (const float*)d_in[8];
    const float* W_ih1   = (const float*)d_in[9];
    const float* b1      = (const float*)d_in[10];
    const float* W_ih2   = (const float*)d_in[11];
    const float* b2      = (const float*)d_in[12];
    const float* b_out   = (const float*)d_in[13];
    const float* hidden0 = (const float*)d_in[14];
    const int*   enc_len = (const int*)d_in[15];
    const int*   y       = (const int*)d_in[16];
    const int*   tf_mask = (const int*)d_in[17];
    float* preds = (float*)d_out;

    // Workspace layout (all 256-B aligned, total ~68.6 MiB)
    char* ws = (char*)d_ws;
    size_t o = 0;
    unsigned short* khL  = (unsigned short*)(ws + o); o += (size_t)B_*NH_*T_*HD_*2; // 33.5MB
    unsigned short* vhL  = (unsigned short*)(ws + o); o += (size_t)B_*NH_*T_*HD_*2; // 33.5MB
    unsigned short* WkB  = (unsigned short*)(ws + o); o += 128*128*2;
    unsigned short* WvB  = (unsigned short*)(ws + o); o += 128*128*2;
    unsigned short* embB = (unsigned short*)(ws + o); o += V_*E_*2;
    unsigned short* W1B  = (unsigned short*)(ws + o); o += (size_t)4*H1_*384*2;
    unsigned short* W2B  = (unsigned short*)(ws + o); o += (size_t)4*D_*H1_*2;
    unsigned short* x1B  = (unsigned short*)(ws + o); o += (size_t)B_*384*2;
    unsigned short* h1B  = (unsigned short*)(ws + o); o += (size_t)B_*H1_*2;
    float* h2f  = (float*)(ws + o); o += (size_t)B_*D_*4;
    float* ctxf = (float*)(ws + o); o += (size_t)B_*D_*4;
    int*   amax = (int*)(ws + o);   o += 512;

    // ---- prep ----
    init_kernel<<<1024, 256, 0, stream>>>(preds, B_*V_*MAXLEN_, hidden0, h2f, amax);
    cvt_kernel<<<64,  256, 0, stream>>>(Wk,    WkB,  128*128);
    cvt_kernel<<<64,  256, 0, stream>>>(Wv,    WvB,  128*128);
    cvt_kernel<<<32,  256, 0, stream>>>(emb,   embB, V_*E_);
    cvt_kernel<<<512, 256, 0, stream>>>(W_ih1, W1B,  4*H1_*384);
    cvt_kernel<<<256, 256, 0, stream>>>(W_ih2, W2B,  4*D_*H1_);
    khvh_kernel<<<(B_*T_)/16, 256, 0, stream>>>(key,   WkB, bk, khL);
    khvh_kernel<<<(B_*T_)/16, 256, 0, stream>>>(value, WvB, bv, vhL);

    // ---- sequential decode: 399 dependent steps, slot s+1 ----
    for (int s = 0; s < MAXLEN_ - 1; ++s) {
        att_kernel<<<B_*NH_, 128, 0, stream>>>(h2f, Wq, bq, khL, vhL, enc_len, ctxf, x1B);
        tok_kernel<<<B_, 64, 0, stream>>>(tf_mask, y, s, amax, embB, x1B);
        g1_kernel<<<dim3(8, 8), 256, 0, stream>>>(x1B, W1B, b1, h1B);
        g2_kernel<<<8, 256, 0, stream>>>(h1B, W2B, b2, h2f);
        out_kernel<<<B_, 32, 0, stream>>>(h2f, ctxf, emb, b_out, preds, s + 1, amax);
    }
}
